// TransformerEncoderLayer_66357244723827
// MI455X (gfx1250) — matrix operand order repaired and checked by reading, not compiled, not run on hardware
//
#include <hip/hip_runtime.h>
#include <math.h>

#define DDIM 128
#define XCW  512

typedef __attribute__((ext_vector_type(16))) __bf16 v16bf;
typedef __attribute__((ext_vector_type(8)))  float  v8f;

// ---------- helpers ----------
__device__ __forceinline__ float wave_sum(float x) {
#pragma unroll
  for (int o = 16; o > 0; o >>= 1) x += __shfl_xor(x, o, 32);
  return x;
}

__device__ __forceinline__ unsigned flipf(float f) {
  unsigned u = __float_as_uint(f);
  return u ^ ((unsigned)((int)u >> 31) | 0x80000000u);
}
__device__ __forceinline__ float unflipf(unsigned u) {
  unsigned v = u ^ ((unsigned)((int)(~u) >> 31) | 0x80000000u);
  return __uint_as_float(v);
}

__device__ __forceinline__ v16bf pack16(float4 a0, float4 a1, float4 a2, float4 a3) {
  v16bf r;
  r[0]=(__bf16)a0.x; r[1]=(__bf16)a0.y; r[2]=(__bf16)a0.z; r[3]=(__bf16)a0.w;
  r[4]=(__bf16)a1.x; r[5]=(__bf16)a1.y; r[6]=(__bf16)a1.z; r[7]=(__bf16)a1.w;
  r[8]=(__bf16)a2.x; r[9]=(__bf16)a2.y; r[10]=(__bf16)a2.z; r[11]=(__bf16)a2.w;
  r[12]=(__bf16)a3.x; r[13]=(__bf16)a3.y; r[14]=(__bf16)a3.z; r[15]=(__bf16)a3.w;
  return r;
}

// ---------- WMMA GEMM: C = act(A[M,K] @ B[K,Nout] + bias) (+resid) ----------
// One wave -> one 16x32 output tile (A fragment reused for two B fragments).
// A fragment (16x32 bf16, ISA 7.12.2): lane m=lane&15, hlf=lane>>4;
//   elem e<8  -> K = kk + 8*hlf + e ; elem e>=8 -> K = kk + 16 + 8*hlf + (e-8)
// B fragment (32x16 bf16): lane n=lane&15 holds column base+n; elem e -> same K order as A.
// C/D: lane n=lane&15; vgpr r -> row r + 8*hlf.
template <int K, bool BIAS, bool RESID, bool GATED, bool RELU>
__global__ void k_gemm(const float* __restrict__ A, int lda,
                       const float* __restrict__ B, int ldb,
                       const float* __restrict__ bias,
                       const float* __restrict__ resid, int ldr,
                       const int* __restrict__ gate,
                       float* __restrict__ C, int ldc,
                       int M, int Nout) {
  int wave = (blockIdx.x * blockDim.x + threadIdx.x) >> 5;
  int lane = threadIdx.x & 31;
  int ntiles = Nout >> 5;                 // 32-wide output tiles
  int mt = wave / ntiles;
  int nt = wave - mt * ntiles;
  if (mt >= (M >> 4)) return;
  int hlf = lane >> 4, mq = lane & 15;
  int row = (mt << 4) + mq;
  bool off = false;
  if (GATED) off = (*gate == 0);
  v8f acc0 = {0.f, 0.f, 0.f, 0.f, 0.f, 0.f, 0.f, 0.f};
  v8f acc1 = acc0;
  if (!off) {
#pragma unroll 4
    for (int kk = 0; kk < K; kk += 32) {
      const float* ap = A + (size_t)row * lda + kk + (hlf << 3);
      __builtin_prefetch(ap + 32, 0, 1);
      float4 a0 = *(const float4*)(ap);
      float4 a1 = *(const float4*)(ap + 4);
      float4 a2 = *(const float4*)(ap + 16);
      float4 a3 = *(const float4*)(ap + 20);
      v16bf af = pack16(a0, a1, a2, a3);
      const float* bp = B + (size_t)(kk + (hlf << 3)) * ldb + (nt << 5) + mq;
      const size_t L = (size_t)ldb;
      float4 b0 = make_float4(bp[0], bp[L], bp[2 * L], bp[3 * L]);
      float4 b1 = make_float4(bp[4 * L], bp[5 * L], bp[6 * L], bp[7 * L]);
      float4 b2 = make_float4(bp[16 * L], bp[17 * L], bp[18 * L], bp[19 * L]);
      float4 b3 = make_float4(bp[20 * L], bp[21 * L], bp[22 * L], bp[23 * L]);
      float4 c0 = make_float4(bp[16], bp[L + 16], bp[2 * L + 16], bp[3 * L + 16]);
      float4 c1 = make_float4(bp[4 * L + 16], bp[5 * L + 16], bp[6 * L + 16], bp[7 * L + 16]);
      float4 c2 = make_float4(bp[16 * L + 16], bp[17 * L + 16], bp[18 * L + 16], bp[19 * L + 16]);
      float4 c3 = make_float4(bp[20 * L + 16], bp[21 * L + 16], bp[22 * L + 16], bp[23 * L + 16]);
      acc0 = __builtin_amdgcn_wmma_f32_16x16x32_bf16(false, af, false, pack16(b0, b1, b2, b3),
                                                     (short)0, acc0, false, false);
      acc1 = __builtin_amdgcn_wmma_f32_16x16x32_bf16(false, af, false, pack16(c0, c1, c2, c3),
                                                     (short)0, acc1, false, false);
    }
  }
  int n = (nt << 5) + mq;
  float bv0 = 0.f, bv1 = 0.f;
  if (BIAS) { if (!off) { bv0 = bias[n]; bv1 = bias[n + 16]; } }
#pragma unroll
  for (int r = 0; r < 8; ++r) {
    int m = (mt << 4) + r + (hlf << 3);
    float v0 = off ? 0.f : acc0[r] + bv0;
    float v1 = off ? 0.f : acc1[r] + bv1;
    if (RESID) {
      v0 += resid[(size_t)m * ldr + n];
      v1 += resid[(size_t)m * ldr + n + 16];
    }
    if (RELU) { v0 = fmaxf(v0, 0.f); v1 = fmaxf(v1, 0.f); }
    C[(size_t)m * ldc + n] = v0;
    C[(size_t)m * ldc + n + 16] = v1;
  }
}

// ---------- elementwise / edge kernels ----------
__global__ void k_fill(float* p, float v, int n) {
  int t = blockIdx.x * blockDim.x + threadIdx.x;
  if (t < n) p[t] = v;
}

__global__ void k_copy_x(const float* __restrict__ x, float* __restrict__ xc, int n) {
  int t = blockIdx.x * blockDim.x + threadIdx.x;
  if (t < n) { int r = t >> 7, c = t & 127; xc[(size_t)r * XCW + c] = x[t]; }
}

__global__ void k_deg(const int* __restrict__ dst, float* deg, int E) {
  int t = blockIdx.x * blockDim.x + threadIdx.x;
  if (t < E) atomicAdd(deg + dst[t], 1.0f);
}

__global__ void k_rsqrt(float* p, int n) {
  int t = blockIdx.x * blockDim.x + threadIdx.x;
  if (t < n) p[t] = rsqrtf(p[t]);
}

// one wave per edge (or self-loop for w >= E); lane covers 4 features
__global__ void k_gcn_agg(const float* __restrict__ h, const int* __restrict__ src,
                          const int* __restrict__ dst, const float* __restrict__ dinv,
                          float* __restrict__ agg, int E, int N) {
  int w = (blockIdx.x * blockDim.x + threadIdx.x) >> 5;
  int lane = threadIdx.x & 31;
  if (w >= E + N) return;
  int s, d;
  if (w < E) { s = src[w]; d = dst[w]; } else { s = d = w - E; }
  float nrm = dinv[s] * dinv[d];
  float4 hv = *(const float4*)(h + (size_t)s * DDIM + lane * 4);
  float* ap = agg + (size_t)d * DDIM + lane * 4;
  atomicAdd(ap + 0, hv.x * nrm);
  atomicAdd(ap + 1, hv.y * nrm);
  atomicAdd(ap + 2, hv.z * nrm);
  atomicAdd(ap + 3, hv.w * nrm);
}

__global__ void k_gcn_fin(const float* __restrict__ agg, const float* __restrict__ b,
                          float* __restrict__ xc, int colOff, int n) {
  int t = blockIdx.x * blockDim.x + threadIdx.x;
  if (t < n) {
    int r = t >> 7, c = t & 127;
    float v = agg[t] + b[c];
    if (v < 0.f) v = 0.f;
    xc[(size_t)r * XCW + colOff + c] = v;
  }
}

__global__ void k_bn_stats(const float* __restrict__ xc, float* mu, float* var, int N) {
  __shared__ float s1[256], s2[256];
  int c = blockIdx.x, tid = threadIdx.x;
  float a = 0.f, b = 0.f;
  for (int r = tid; r < N; r += 256) {
    float v = xc[(size_t)r * XCW + c];
    a += v; b += v * v;
  }
  s1[tid] = a; s2[tid] = b; __syncthreads();
  for (int o = 128; o > 0; o >>= 1) {
    if (tid < o) { s1[tid] += s1[tid + o]; s2[tid] += s2[tid + o]; }
    __syncthreads();
  }
  if (tid == 0) {
    float m = s1[0] / (float)N;
    mu[c] = m;
    var[c] = s2[0] / (float)N - m * m;
  }
}

__global__ void k_bn_norm(float* __restrict__ xc, const float* __restrict__ mu,
                          const float* __restrict__ var, const float* __restrict__ g,
                          const float* __restrict__ b, int n) {
  int t = blockIdx.x * blockDim.x + threadIdx.x;
  if (t < n) {
    int c = t & (XCW - 1);
    xc[t] = (xc[t] - mu[c]) * rsqrtf(var[c] + 1e-5f) * g[c] + b[c];
  }
}

// wave per edge; lanes: head = lane>>2, 4 lanes x 4 elems = 16-dim head dot
__global__ void k_logits(const float* __restrict__ qk, const int* __restrict__ src,
                         const int* __restrict__ dst, float* __restrict__ logits,
                         unsigned* __restrict__ mU, int E) {
  int w = (blockIdx.x * blockDim.x + threadIdx.x) >> 5;
  int lane = threadIdx.x & 31;
  if (w >= E) return;
  int j = src[w], i = dst[w];
  int head = lane >> 2, p = lane & 3;
  float4 qv = *(const float4*)(qk + (size_t)i * 256 + 128 + head * 16 + p * 4);
  float4 kv = *(const float4*)(qk + (size_t)j * 256 + head * 16 + p * 4);
  float s = qv.x * kv.x + qv.y * kv.y + qv.z * kv.z + qv.w * kv.w;
  s += __shfl_xor(s, 1, 32);
  s += __shfl_xor(s, 2, 32);
  if (p == 0) {
    float l = s * 0.25f;  // DH^-0.5, DH=16
    logits[(size_t)w * 8 + head] = l;
    atomicMax(mU + (size_t)i * 8 + head, flipf(l));
  }
}

__global__ void k_mfix(float* m, int n) {
  int t = blockIdx.x * blockDim.x + threadIdx.x;
  if (t < n) {
    float f = unflipf(__float_as_uint(m[t]));
    m[t] = __builtin_isfinite(f) ? f : 0.0f;
  }
}

__global__ void k_exp(float* __restrict__ lg, const float* __restrict__ m,
                      const int* __restrict__ dst, float* __restrict__ denom, int n) {
  int t = blockIdx.x * blockDim.x + threadIdx.x;
  if (t < n) {
    int e = t >> 3, h = t & 7;
    int i = dst[e];
    float ee = __expf(lg[t] - m[(size_t)i * 8 + h]);
    lg[t] = ee;
    atomicAdd(denom + (size_t)i * 8 + h, ee);
  }
}

__global__ void k_attnagg(const float* __restrict__ ee, const float* __restrict__ denom,
                          const float* __restrict__ v, const int* __restrict__ src,
                          const int* __restrict__ dst, float* __restrict__ agg, int E) {
  int w = (blockIdx.x * blockDim.x + threadIdx.x) >> 5;
  int lane = threadIdx.x & 31;
  if (w >= E) return;
  int j = src[w], i = dst[w];
  int head = lane >> 2;
  float alpha = ee[(size_t)w * 8 + head] / (denom[(size_t)i * 8 + head] + 1e-16f);
  float4 vv = *(const float4*)(v + (size_t)j * DDIM + lane * 4);
  float* ap = agg + (size_t)i * DDIM + lane * 4;
  atomicAdd(ap + 0, vv.x * alpha);
  atomicAdd(ap + 1, vv.y * alpha);
  atomicAdd(ap + 2, vv.z * alpha);
  atomicAdd(ap + 3, vv.w * alpha);
}

// wave per row LayerNorm over 128 features
__global__ void k_ln(const float* __restrict__ in, const float* __restrict__ g,
                     const float* __restrict__ b, float* __restrict__ out, int N) {
  int w = (blockIdx.x * blockDim.x + threadIdx.x) >> 5;
  int lane = threadIdx.x & 31;
  if (w >= N) return;
  float4 v = *(const float4*)(in + (size_t)w * DDIM + lane * 4);
  float mu = wave_sum(v.x + v.y + v.z + v.w) * (1.0f / 128.0f);
  float dx = v.x - mu, dy = v.y - mu, dz = v.z - mu, dw = v.w - mu;
  float var = wave_sum(dx * dx + dy * dy + dz * dz + dw * dw) * (1.0f / 128.0f);
  float rs = rsqrtf(var + 1e-5f);
  int f = lane * 4;
  float* op = out + (size_t)w * DDIM + f;
  op[0] = dx * rs * g[f + 0] + b[f + 0];
  op[1] = dy * rs * g[f + 1] + b[f + 1];
  op[2] = dz * rs * g[f + 2] + b[f + 2];
  op[3] = dw * rs * g[f + 3] + b[f + 3];
}

// ---------- host side ----------
static inline int gemm_blocks(int M, int Nout) {
  int tiles = (M / 16) * (Nout / 32);
  return (tiles + 7) / 8;
}

static inline void fill(hipStream_t s, float* p, float v, int n) {
  k_fill<<<(n + 255) / 256, 256, 0, s>>>(p, v, n);
}

extern "C" void kernel_launch(void* const* d_in, const int* in_sizes, int n_in,
                              void* d_out, int out_size, void* d_ws, size_t ws_size,
                              hipStream_t stream) {
  const float* x   = (const float*)d_in[0];
  const int*   ei  = (const int*)d_in[1];
  const int*   dag = (const int*)d_in[2];
  const int*   SAT = (const int*)d_in[3];
  const float* Wg[3]  = {(const float*)d_in[4], (const float*)d_in[6], (const float*)d_in[8]};
  const float* bg[3]  = {(const float*)d_in[5], (const float*)d_in[7], (const float*)d_in[9]};
  const float* bn_g = (const float*)d_in[10];
  const float* bn_b = (const float*)d_in[11];
  const float* Wse  = (const float*)d_in[12];
  const float* bse  = (const float*)d_in[13];
  const float* Wqk  = (const float*)d_in[14];
  const float* Wv   = (const float*)d_in[15];
  const float* Wo   = (const float*)d_in[16];
  const float* bo   = (const float*)d_in[17];
  const float* ln1g = (const float*)d_in[18];
  const float* ln1b = (const float*)d_in[19];
  const float* W1   = (const float*)d_in[20];
  const float* b1   = (const float*)d_in[21];
  const float* W2   = (const float*)d_in[22];
  const float* b2   = (const float*)d_in[23];
  const float* ln2g = (const float*)d_in[24];
  const float* ln2b = (const float*)d_in[25];

  const int N = in_sizes[0] / DDIM;    // 50000
  const int E = in_sizes[1] / 2;       // 800000
  const int* src_g = ei,  *dst_g = ei + E;
  const int* src_a = dag, *dst_a = dag + E;

  float* ws = (float*)d_ws;
  size_t off = 0;
  auto take = [&](size_t n) { float* p = ws + off; off += (n + 63) & ~(size_t)63; return p; };
  float* xc   = take((size_t)N * XCW);   // concat / later FFN hidden
  float* bufA = take((size_t)N * DDIM);  // gcn gemm out / x_att / attn agg
  float* bufB = take((size_t)N * DDIM);  // gcn agg / v / ffn2 out
  float* qkb  = take((size_t)N * 256);   // qk / later hpre+hbuf
  float* lgb  = take((size_t)E * 8);     // logits -> exp weights
  float* mb   = take((size_t)N * 8);     // segment max (flipped-uint then float)
  float* dnb  = take((size_t)N * 8);     // softmax denom
  float* degb = take((size_t)N);
  float* mub  = take(512);
  float* varb = take(512);
  (void)ws_size; (void)n_in; (void)out_size;

  const int NE = N * DDIM;      // 6.4M
  const int eW = (E + 7) / 8;   // blocks for wave-per-edge kernels

  // ---- structure extractor (always computed; SAT gates the add) ----
  k_copy_x<<<(NE + 255) / 256, 256, 0, stream>>>(x, xc, NE);
  fill(stream, degb, 1.0f, N);  // self-loop
  k_deg<<<(E + 255) / 256, 256, 0, stream>>>(dst_g, degb, E);
  k_rsqrt<<<(N + 255) / 256, 256, 0, stream>>>(degb, N);

  for (int L = 0; L < 3; ++L) {
    k_gemm<128, false, false, false, false><<<gemm_blocks(N, DDIM), 256, 0, stream>>>(
        xc + L * DDIM, XCW, Wg[L], DDIM, nullptr, nullptr, 0, nullptr, bufA, DDIM, N, DDIM);
    fill(stream, bufB, 0.0f, NE);
    k_gcn_agg<<<(E + N + 7) / 8, 256, 0, stream>>>(bufA, src_g, dst_g, degb, bufB, E, N);
    k_gcn_fin<<<(NE + 255) / 256, 256, 0, stream>>>(bufB, bg[L], xc, (L + 1) * DDIM, NE);
  }
  k_bn_stats<<<XCW, 256, 0, stream>>>(xc, mub, varb, N);
  k_bn_norm<<<((N * XCW) + 255) / 256, 256, 0, stream>>>(xc, mub, varb, bn_g, bn_b, N * XCW);
  // x_att = x + (xn @ Wse + bse)   (gated by SAT -> x_att = x)
  k_gemm<512, true, true, true, false><<<gemm_blocks(N, DDIM), 256, 0, stream>>>(
      xc, XCW, Wse, DDIM, bse, x, DDIM, SAT, bufA, DDIM, N, DDIM);

  // ---- attention ----
  k_gemm<128, false, false, false, false><<<gemm_blocks(N, 256), 256, 0, stream>>>(
      bufA, DDIM, Wqk, 256, nullptr, nullptr, 0, nullptr, qkb, 256, N, 256);
  k_gemm<128, false, false, false, false><<<gemm_blocks(N, DDIM), 256, 0, stream>>>(
      x, DDIM, Wv, DDIM, nullptr, nullptr, 0, nullptr, bufB, DDIM, N, DDIM);
  fill(stream, mb, 0.0f, N * 8);     // bits==0 == flipped-min identity for max
  fill(stream, dnb, 0.0f, N * 8);
  k_logits<<<eW, 256, 0, stream>>>(qkb, src_a, dst_a, lgb, (unsigned*)mb, E);
  k_mfix<<<((N * 8) + 255) / 256, 256, 0, stream>>>(mb, N * 8);
  k_exp<<<((E * 8) + 255) / 256, 256, 0, stream>>>(lgb, mb, dst_a, dnb, E * 8);
  fill(stream, bufA, 0.0f, NE);      // x_att no longer needed
  k_attnagg<<<eW, 256, 0, stream>>>(lgb, dnb, bufB, src_a, dst_a, bufA, E);

  // ---- output projection + LN1 ----
  float* hpre = qkb;                 // qk consumed
  float* hbuf = qkb + (size_t)N * DDIM;
  k_gemm<128, true, true, false, false><<<gemm_blocks(N, DDIM), 256, 0, stream>>>(
      bufA, DDIM, Wo, DDIM, bo, x, DDIM, nullptr, hpre, DDIM, N, DDIM);
  k_ln<<<(N + 7) / 8, 256, 0, stream>>>(hpre, ln1g, ln1b, hbuf, N);

  // ---- FFN + LN2 ----
  k_gemm<128, true, false, false, true><<<gemm_blocks(N, XCW), 256, 0, stream>>>(
      hbuf, DDIM, W1, XCW, b1, nullptr, 0, nullptr, xc, XCW, N, XCW);
  k_gemm<512, true, true, false, false><<<gemm_blocks(N, DDIM), 256, 0, stream>>>(
      xc, XCW, W2, DDIM, b2, hbuf, DDIM, nullptr, bufB, DDIM, N, DDIM);
  k_ln<<<(N + 7) / 8, 256, 0, stream>>>(bufB, ln2g, ln2b, (float*)d_out, N);
}
